// QuantizedLinear_42099269435592
// MI455X (gfx1250) — matrix operand order repaired and checked by reading, not compiled, not run on hardware
//
#include <hip/hip_runtime.h>

// ---------------------------------------------------------------------------
// QuantizedLinear: out = qdq( x @ ((w_i8 - zp_w)*s_w)^T + bias )
// Plan: pass1 convert x fp32->fp16 (+ per-row sums), pass2 convert w int32->fp16,
// pass3 WMMA f16 GEMM (128x128 tile / 256 thr / 8 waves) with double-buffered
// LDS staged by async global->LDS copies, fused quant-dequant epilogue.
// LDS tiles use a padded row stride (72 halves = 36 banks) so all ds_load_b128
// fragment reads are bank-conflict-free (gcd(36,64)=4 tiles the 16 bank-groups).
// ---------------------------------------------------------------------------

#define TOKENS 8192
#define D_IN   4096
#define D_OUT  4096

#define BM 128
#define BN 128
#define BK 64   // K-block (halves)
#define BKP 72  // padded LDS row stride in halves (144 B = 36 banks)

typedef __attribute__((ext_vector_type(16))) _Float16 v16h;
typedef __attribute__((ext_vector_type(8)))  _Float16 v8h;
typedef __attribute__((ext_vector_type(4)))  _Float16 v4h;
typedef __attribute__((ext_vector_type(8)))  float    v8f;
typedef __attribute__((ext_vector_type(4)))  int      v4i;

// Address-space-qualified pointer types for the async-copy builtin:
// param0 is int4 in AS(1) (global), param1 is int4 in AS(3) (LDS).
typedef __attribute__((address_space(1))) v4i* gv4i_p;
typedef __attribute__((address_space(3))) v4i* lv4i_p;

#if defined(__gfx1250__) && __has_builtin(__builtin_amdgcn_global_load_async_to_lds_b128)
#define USE_ASYNC 1
#else
#define USE_ASYNC 0
#endif

__device__ __forceinline__ void wait_async_done() {
#if USE_ASYNC
#if __has_builtin(__builtin_amdgcn_s_wait_asynccnt)
  __builtin_amdgcn_s_wait_asynccnt(0);
#else
  asm volatile("s_wait_asynccnt 0" ::: "memory");
#endif
#endif
}

// A fragment 16x32 f16 (rows m0..m0+15, cols kk..kk+31) per ISA layout:
// lane<16: row=lane, K {0..7,16..23}; lane>=16: row=(lane&15), K {8..15,24..31}
__device__ __forceinline__ v16h load_frag_a(const _Float16* t, int m0, int kk, int lane) {
  const int r  = lane & 15;
  const int kh = (lane >> 4) << 3;               // 0 or 8
  const _Float16* p = t + (m0 + r) * BKP + kk + kh;
  v8h lo = *(const v8h*)p;                        // K = kk+kh+0..7
  v8h hi = *(const v8h*)(p + 16);                 // K = kk+kh+16..23
  v16h o;
#pragma unroll
  for (int i = 0; i < 8; ++i) { o[i] = lo[i]; o[i + 8] = hi[i]; }
  return o;
}

// B fragment 32x16 f16 (cols n0..n0+15): lane<16 holds column n=lane, K {0..7,16..23};
// lane>=16 holds same column, K {8..15,24..31}. W tile is [N][BKP] row-major.
__device__ __forceinline__ v16h load_frag_b(const _Float16* t, int n0, int kk, int lane) {
  const _Float16* p = t + (n0 + (lane & 15)) * BKP + kk + ((lane >> 4) << 3);
  v8h lo = *(const v8h*)p;
  v8h hi = *(const v8h*)(p + 16);
  v16h o;
#pragma unroll
  for (int i = 0; i < 8; ++i) { o[i] = lo[i]; o[i + 8] = hi[i]; }
  return o;
}

// ---------------------------------------------------------------------------
// Pass 1: x fp32 -> fp16, plus row sums (for weight zero-point correction).
// One block per token row; 256 thr * 16 elems.
// ---------------------------------------------------------------------------
__global__ __launch_bounds__(256) void xcvt_kernel(const float* __restrict__ x,
                                                   _Float16* __restrict__ xh,
                                                   float* __restrict__ rowsum) {
  const int row = blockIdx.x;
  const int tid = threadIdx.x;
  const float*    xr = x  + (size_t)row * D_IN + tid * 16;
  _Float16*       xo = xh + (size_t)row * D_IN + tid * 16;
  float s = 0.0f;
#pragma unroll
  for (int q = 0; q < 4; ++q) {
    float4 v = *(const float4*)(xr + q * 4);
    s += v.x + v.y + v.z + v.w;
    v4h h = { (_Float16)v.x, (_Float16)v.y, (_Float16)v.z, (_Float16)v.w };
    *(v4h*)(xo + q * 4) = h;
  }
  __shared__ float red[256];
  red[tid] = s;
  __syncthreads();
#pragma unroll
  for (int off = 128; off > 0; off >>= 1) {
    if (tid < off) red[tid] += red[tid + off];
    __syncthreads();
  }
  if (tid == 0) rowsum[row] = red[0];
}

// ---------------------------------------------------------------------------
// Pass 2: weight int32 (int8 values) -> fp16 (exact: |w| <= 128).
// ---------------------------------------------------------------------------
__global__ __launch_bounds__(256) void wcvt_kernel(const int* __restrict__ w,
                                                   _Float16* __restrict__ wh) {
  const size_t i = ((size_t)blockIdx.x * 256 + threadIdx.x) * 4;
  int4 v = *(const int4*)(w + i);
  v4h h = { (_Float16)v.x, (_Float16)v.y, (_Float16)v.z, (_Float16)v.w };
  *(v4h*)(wh + i) = h;
}

// ---------------------------------------------------------------------------
// Pass 3: WMMA GEMM + quant-dequant epilogue.
// Block = 128x128 output, 8 waves in 2(M)x4(N); each wave 64x32 = 4x2 WMMA accs.
// ---------------------------------------------------------------------------
__global__ __launch_bounds__(256) void gemm_kernel(
    const _Float16* __restrict__ Xh, const _Float16* __restrict__ Wh,
    const float* __restrict__ rowsum, const float* __restrict__ bias,
    const float* __restrict__ wscale_p, const int* __restrict__ wzp_p,
    const float* __restrict__ oscale_p, const int* __restrict__ ozp_p,
    float* __restrict__ out) {
  __shared__ __align__(32) _Float16 ldsX[2][BM * BKP];   // 2 * 18 KB
  __shared__ __align__(32) _Float16 ldsW[2][BN * BKP];   // 2 * 18 KB

  const int tid  = threadIdx.x;
  const int lane = tid & 31;
  const int wave = tid >> 5;
  const int wm = (wave >> 2) * 64;   // wave M offset in tile
  const int wn = (wave & 3) * 32;    // wave N offset in tile
  const int m_wg = blockIdx.y * BM;
  const int n_wg = blockIdx.x * BN;

  const float wscale = wscale_p[0];
  const float wzp    = (float)wzp_p[0];
  const float oscale = oscale_p[0];
  const float ozp    = (float)ozp_p[0];

  const v8f vzero = {};
  v8f acc[4][2];
#pragma unroll
  for (int i = 0; i < 4; ++i)
#pragma unroll
    for (int j = 0; j < 2; ++j) acc[i][j] = vzero;

  // Stage one 128x64 X tile + 128x64 W tile into LDS buffer `buf`.
  auto stage = [&](int buf, int k0) {
#pragma unroll
    for (int q = 0; q < 4; ++q) {
      const int c   = tid + q * 256;     // 1024 16B chunks per tile
      const int row = c >> 3;
      const int col = (c & 7) << 3;      // halves
      const _Float16* gx = Xh + (size_t)(m_wg + row) * D_IN + k0 + col;
      const _Float16* gw = Wh + (size_t)(n_wg + row) * D_IN + k0 + col;
      _Float16* lx = &ldsX[buf][row * BKP + col];
      _Float16* lw = &ldsW[buf][row * BKP + col];
#if USE_ASYNC
      __builtin_amdgcn_global_load_async_to_lds_b128((gv4i_p)gx, (lv4i_p)lx, 0, 0);
      __builtin_amdgcn_global_load_async_to_lds_b128((gv4i_p)gw, (lv4i_p)lw, 0, 0);
#else
      *(uint4*)lx = *(const uint4*)gx;
      *(uint4*)lw = *(const uint4*)gw;
#endif
    }
  };

  stage(0, 0);
  wait_async_done();
  __syncthreads();

  const int nk = D_IN / BK;   // 64 K-steps
  for (int kt = 0; kt < nk; ++kt) {
    const int buf = kt & 1;
    if (kt + 1 < nk) stage(buf ^ 1, (kt + 1) * BK);   // overlap with compute

#pragma unroll
    for (int kk = 0; kk < BK; kk += 32) {
      v16h af[4], bf[2];
#pragma unroll
      for (int i = 0; i < 4; ++i) af[i] = load_frag_a(&ldsX[buf][0], wm + i * 16, kk, lane);
#pragma unroll
      for (int j = 0; j < 2; ++j) bf[j] = load_frag_b(&ldsW[buf][0], wn + j * 16, kk, lane);
#pragma unroll
      for (int i = 0; i < 4; ++i)
#pragma unroll
        for (int j = 0; j < 2; ++j)
          acc[i][j] = __builtin_amdgcn_wmma_f32_16x16x32_f16(
              false, af[i], false, bf[j], (short)0, acc[i][j], false, false);
    }

    wait_async_done();   // next buffer landed
    __syncthreads();     // everyone done reading current buffer
  }

  // Epilogue: zp correction, scale, bias, quantize->clamp->dequantize.
  const float inv_os = 1.0f / oscale;
#pragma unroll
  for (int i = 0; i < 4; ++i) {
    const int mbase = m_wg + wm + i * 16 + ((lane >> 4) << 3);  // +8 for hi half-wave
#pragma unroll
    for (int j = 0; j < 2; ++j) {
      const int n = n_wg + wn + j * 16 + (lane & 15);
      const float bn = bias[n];
#pragma unroll
      for (int r = 0; r < 8; ++r) {
        const int m = mbase + r;
        const float val = (acc[i][j][r] - wzp * rowsum[m]) * wscale + bn;
        float q = rintf(val * inv_os) + ozp;
        q = fminf(fmaxf(q, 0.0f), 255.0f);
        out[(size_t)m * D_OUT + n] = (q - ozp) * oscale;
      }
    }
  }
}

// ---------------------------------------------------------------------------
extern "C" void kernel_launch(void* const* d_in, const int* in_sizes, int n_in,
                              void* d_out, int out_size, void* d_ws, size_t ws_size,
                              hipStream_t stream) {
  const float* x      = (const float*)d_in[0];
  const int*   w8     = (const int*)d_in[1];
  const float* wscale = (const float*)d_in[2];
  const int*   wzp    = (const int*)d_in[3];
  const float* bias   = (const float*)d_in[4];
  const float* oscale = (const float*)d_in[5];
  const int*   ozp    = (const int*)d_in[6];
  float* out = (float*)d_out;

  // Workspace layout: Xh fp16 (64 MB) | Wh fp16 (32 MB) | rowsum f32 (32 KB)
  _Float16* Xh = (_Float16*)d_ws;
  _Float16* Wh = (_Float16*)((char*)d_ws + (size_t)TOKENS * D_IN * sizeof(_Float16));
  float* rowsum = (float*)((char*)d_ws + (size_t)TOKENS * D_IN * sizeof(_Float16)
                                       + (size_t)D_OUT * D_IN * sizeof(_Float16));

  xcvt_kernel<<<TOKENS, 256, 0, stream>>>(x, Xh, rowsum);
  wcvt_kernel<<<(unsigned)(((size_t)D_OUT * D_IN) / (256 * 4)), 256, 0, stream>>>(w8, Wh);

  dim3 grid(D_OUT / BN, TOKENS / BM);   // 32 x 64 = 2048 workgroups
  gemm_kernel<<<grid, 256, 0, stream>>>(Xh, Wh, rowsum, bias,
                                        wscale, wzp, oscale, ozp, out);
}
